// MultiheadAttention_33698313404919
// MI455X (gfx1250) — matrix operand order repaired and checked by reading, not compiled, not run on hardware
//
#include <hip/hip_runtime.h>
#include <hip/hip_bf16.h>

typedef _Float16 v16h __attribute__((ext_vector_type(16)));
typedef _Float16 h8   __attribute__((ext_vector_type(8)));
typedef float    v8f  __attribute__((ext_vector_type(8)));

constexpr int Bb = 4, Ss = 2048, DM = 1024, Hh = 16, DEPTH = 64;
constexpr int Kdim = DM;                 // GEMM reduction dim for projections
constexpr float SCALE = 0.125f;          // 1/sqrt(64)

// ---- fragment load helpers (contiguous 8-half groups -> b128 loads) ----
static __device__ inline v16h ld16h(const _Float16* p0, const _Float16* p1) {
  h8 a = *(const h8*)p0;
  h8 b = *(const h8*)p1;
  v16h r;
#pragma unroll
  for (int i = 0; i < 8; ++i) { r[i] = a[i]; r[i + 8] = b[i]; }
  return r;
}
static __device__ inline v16h ld16h_f32(const float* p0, const float* p1) {
  v16h r;
#pragma unroll
  for (int i = 0; i < 8; ++i) { r[i] = (_Float16)p0[i]; r[i + 8] = (_Float16)p1[i]; }
  return r;
}
static __device__ inline v8f wmma16(v16h a, v16h b, v8f c) {
  return __builtin_amdgcn_wmma_f32_16x16x32_f16(false, a, false, b, (short)0, c,
                                                false, false);
}

// ---- weight f32 -> f16 conversion ----
__global__ void cvt_f32_f16(const float* __restrict__ in, _Float16* __restrict__ out,
                            int n) {
  int i = blockIdx.x * blockDim.x + threadIdx.x;
  int stride = gridDim.x * blockDim.x;
  for (; i < n; i += stride) out[i] = (_Float16)in[i];
}

// ---- projection GEMM: Y = X @ W^T + b ; one wave -> 16x64 tile ----
// MODE 0: X f32, out f16 head layout [B,H,S,64]       (Q, K)
// MODE 1: X f32, out f16 transposed head [B,H,64,S]   (V)
// MODE 2: X f16 (ctx), out f32 flat [B*S, D]          (output projection)
template <int MODE>
__global__ __launch_bounds__(256) void proj_gemm(
    const float* __restrict__ Xf, const _Float16* __restrict__ Xh,
    const _Float16* __restrict__ W, const float* __restrict__ bias,
    float* __restrict__ OutF, _Float16* __restrict__ OutH) {
  const int lane = threadIdx.x & 31;
  const int wid = threadIdx.x >> 5;
  const int l16 = lane & 15;
  const int hi = lane >> 4;
  const int mt = blockIdx.x * 8 + wid;  // 512 M-tiles
  const int n0 = blockIdx.y * 64;       // 16 N-chunks
  const int m0 = mt * 16;
  const int arow = m0 + l16;

  v8f acc[4] = {};
  for (int kb = 0; kb < Kdim; kb += 32) {
    v16h a;
    if (MODE == 2) {
      const _Float16* p = Xh + (size_t)arow * Kdim + kb + hi * 8;
      __builtin_prefetch(p + 32, 0, 1);
      a = ld16h(p, p + 16);
    } else {
      const float* p = Xf + (size_t)arow * Kdim + kb + hi * 8;
      __builtin_prefetch(p + 32, 0, 1);
      a = ld16h_f32(p, p + 16);
    }
#pragma unroll
    for (int c = 0; c < 4; ++c) {
      const _Float16* bp = W + (size_t)(n0 + c * 16 + l16) * Kdim + kb + hi * 8;
      acc[c] = wmma16(a, ld16h(bp, bp + 16), acc[c]);
    }
  }
#pragma unroll
  for (int c = 0; c < 4; ++c) {
    const int n = n0 + c * 16 + l16;
    const float bv = bias[n];
#pragma unroll
    for (int r = 0; r < 8; ++r) {
      const int mrow = m0 + r + hi * 8;
      const float v = acc[c][r] + bv;
      if (MODE == 2) {
        OutF[(size_t)mrow * DM + n] = v;
      } else {
        const int hh = n >> 6, dd = n & 63;
        const int bb = mrow >> 11, ss = mrow & (Ss - 1);
        if (MODE == 0)
          OutH[((size_t)(bb * Hh + hh) * Ss + ss) * DEPTH + dd] = (_Float16)v;
        else
          OutH[((size_t)(bb * Hh + hh) * DEPTH + dd) * Ss + ss] = (_Float16)v;
      }
    }
  }
}

// ---- scores + causal mask + softmax -> attn probs (f32, written once) ----
// One workgroup (8 waves) per 16-row q-block of one (b,h).
// Three sweeps over the (recomputed) logit tiles:
//   1) row max      : 1 fmax/logit, no exps, no dependency chain
//   2) row sum-exp  : 1 exp/logit, independent adds
//   3) normalize+store (the HBM-bound part: 1.07 GB of attn probs)
__global__ __launch_bounds__(256) void attn_scores(const _Float16* __restrict__ Qh,
                                                   const _Float16* __restrict__ Kh,
                                                   float* __restrict__ attn) {
  const int qt = blockIdx.x, h = blockIdx.y, b = blockIdx.z;
  const int lane = threadIdx.x & 31, wid = threadIdx.x >> 5;
  const int l16 = lane & 15, hiL = lane >> 4;
  const int q0 = qt * 16;
  const size_t headoff = (size_t)(b * Hh + h) * Ss * DEPTH;
  const _Float16* Q = Qh + headoff;
  const _Float16* Kp = Kh + headoff;
  float* A = attn + (size_t)(b * Hh + h) * Ss * Ss;

  __shared__ float red[8][16];   // per-wave partials (max, then sum)
  __shared__ float Mrow[16], Lrow[16];

  // Q fragments (reused for every key tile): depth 0..31 and 32..63
  const _Float16* qr = Q + (size_t)(q0 + l16) * DEPTH;
  const v16h a0 = ld16h(qr + hiL * 8, qr + hiL * 8 + 16);
  const v16h a1 = ld16h(qr + 32 + hiL * 8, qr + 48 + hiL * 8);

  // ---- sweep 1: per-lane row max over causally-live tiles (nt <= qt) ----
  float mrun[8];
#pragma unroll
  for (int r = 0; r < 8; ++r) mrun[r] = -3.0e38f;
  for (int nt = wid; nt <= qt; nt += 8) {
    const _Float16* kr = Kp + (size_t)(nt * 16 + l16) * DEPTH + hiL * 8;
    __builtin_prefetch(kr + 8 * 16 * DEPTH, 0, 1);  // this wave's next key tile
    v8f c = {};
    c = wmma16(a0, ld16h(kr, kr + 16), c);
    c = wmma16(a1, ld16h(kr + 32, kr + 48), c);
    const int j = nt * 16 + l16;
#pragma unroll
    for (int r = 0; r < 8; ++r) {
      const int irow = q0 + r + hiL * 8;
      const float v = c[r] * SCALE;
      if (j <= irow) mrun[r] = fmaxf(mrun[r], v);
    }
  }
#pragma unroll
  for (int r = 0; r < 8; ++r)
#pragma unroll
    for (int off = 1; off < 16; off <<= 1)
      mrun[r] = fmaxf(mrun[r], __shfl_xor(mrun[r], off, 32));
  if (l16 == 0)
#pragma unroll
    for (int r = 0; r < 8; ++r) red[wid][r + hiL * 8] = mrun[r];
  __syncthreads();
  if (threadIdx.x < 16) {
    float Mv = -3.0e38f;
    for (int w = 0; w < 8; ++w) Mv = fmaxf(Mv, red[w][threadIdx.x]);
    Mrow[threadIdx.x] = Mv;
  }
  __syncthreads();
  float mfin[8];
#pragma unroll
  for (int r = 0; r < 8; ++r) mfin[r] = Mrow[r + hiL * 8];
  __syncthreads();  // red[] about to be reused for the sum

  // ---- sweep 2: per-lane row sum of exp(v - M) ----
  float lrun[8] = {};
  for (int nt = wid; nt <= qt; nt += 8) {
    const _Float16* kr = Kp + (size_t)(nt * 16 + l16) * DEPTH + hiL * 8;
    v8f c = {};
    c = wmma16(a0, ld16h(kr, kr + 16), c);
    c = wmma16(a1, ld16h(kr + 32, kr + 48), c);
    const int j = nt * 16 + l16;
#pragma unroll
    for (int r = 0; r < 8; ++r) {
      const int irow = q0 + r + hiL * 8;
      const float e = __expf(c[r] * SCALE - mfin[r]);
      lrun[r] += (j <= irow) ? e : 0.0f;
    }
  }
#pragma unroll
  for (int r = 0; r < 8; ++r)
#pragma unroll
    for (int off = 1; off < 16; off <<= 1)
      lrun[r] += __shfl_xor(lrun[r], off, 32);
  if (l16 == 0)
#pragma unroll
    for (int r = 0; r < 8; ++r) red[wid][r + hiL * 8] = lrun[r];
  __syncthreads();
  if (threadIdx.x < 16) {
    float Lv = 0.0f;
    for (int w = 0; w < 8; ++w) Lv += red[w][threadIdx.x];
    Lrow[threadIdx.x] = 1.0f / Lv;
  }
  __syncthreads();
  float lfin[8];
#pragma unroll
  for (int r = 0; r < 8; ++r) lfin[r] = Lrow[r + hiL * 8];

  // ---- sweep 3: normalize + store probs; fully-masked tiles store zeros ----
  for (int nt = wid; nt < Ss / 16; nt += 8) {
    const int j = nt * 16 + l16;
    if (nt > qt) {
#pragma unroll
      for (int r = 0; r < 8; ++r) A[(size_t)(q0 + r + hiL * 8) * Ss + j] = 0.0f;
      continue;
    }
    const _Float16* kr = Kp + (size_t)(nt * 16 + l16) * DEPTH + hiL * 8;
    v8f c = {};
    c = wmma16(a0, ld16h(kr, kr + 16), c);
    c = wmma16(a1, ld16h(kr + 32, kr + 48), c);
#pragma unroll
    for (int r = 0; r < 8; ++r) {
      const int irow = q0 + r + hiL * 8;
      const float p =
          (j > irow) ? 0.0f : __expf(c[r] * SCALE - mfin[r]) * lfin[r];
      A[(size_t)irow * Ss + j] = p;
    }
  }
}

// ---- ctx = attn @ V : one wave -> 16x64 tile; skips zero (masked) k-tiles ----
__global__ __launch_bounds__(256) void attn_ctx(const float* __restrict__ attn,
                                                const _Float16* __restrict__ VhT,
                                                _Float16* __restrict__ ctx) {
  const int lane = threadIdx.x & 31, wid = threadIdx.x >> 5;
  const int l16 = lane & 15, hiL = lane >> 4;
  const int qt = blockIdx.x * 8 + wid;
  const int h = blockIdx.y, b = blockIdx.z;
  const int q0 = qt * 16;
  const float* A = attn + (size_t)(b * Hh + h) * Ss * Ss;
  const _Float16* Vt = VhT + (size_t)(b * Hh + h) * DEPTH * Ss;  // [64][S]

  v8f acc[4] = {};
  for (int kb = 0; kb < q0 + 16; kb += 32) {  // causal: only lower-tri tiles
    const float* ar = A + (size_t)(q0 + l16) * Ss + kb + hiL * 8;
    __builtin_prefetch(ar + 32, 0, 1);
    v16h a = ld16h_f32(ar, ar + 16);
#pragma unroll
    for (int c = 0; c < 4; ++c) {
      const _Float16* bp = Vt + (size_t)(c * 16 + l16) * Ss + kb + hiL * 8;
      acc[c] = wmma16(a, ld16h(bp, bp + 16), acc[c]);
    }
  }
#pragma unroll
  for (int c = 0; c < 4; ++c) {
#pragma unroll
    for (int r = 0; r < 8; ++r) {
      const int srow = q0 + r + hiL * 8;
      ctx[(size_t)(b * Ss + srow) * DM + h * DEPTH + c * 16 + l16] =
          (_Float16)acc[c][r];
    }
  }
}

extern "C" void kernel_launch(void* const* d_in, const int* in_sizes, int n_in,
                              void* d_out, int out_size, void* d_ws, size_t ws_size,
                              hipStream_t stream) {
  (void)in_sizes; (void)n_in; (void)out_size; (void)ws_size;
  const float* q   = (const float*)d_in[0];
  const float* k   = (const float*)d_in[1];
  const float* v   = (const float*)d_in[2];
  // d_in[3] = mask (causality implemented analytically)
  const float* wq  = (const float*)d_in[4];
  const float* wqb = (const float*)d_in[5];
  const float* wk  = (const float*)d_in[6];
  const float* wkb = (const float*)d_in[7];
  const float* wv  = (const float*)d_in[8];
  const float* wvb = (const float*)d_in[9];
  const float* wo  = (const float*)d_in[10];
  const float* wob = (const float*)d_in[11];

  float* out  = (float*)d_out;                       // [B,S,D]
  float* attn = out + (size_t)Bb * Ss * DM;          // [B,H,S,S]

  char* ws = (char*)d_ws;
  const size_t wbytes = (size_t)DM * DM * sizeof(_Float16);                // 2 MB
  const size_t hbytes = (size_t)Bb * Hh * Ss * DEPTH * sizeof(_Float16);   // 16 MB
  _Float16* wq16 = (_Float16*)ws; ws += wbytes;
  _Float16* wk16 = (_Float16*)ws; ws += wbytes;
  _Float16* wv16 = (_Float16*)ws; ws += wbytes;
  _Float16* wo16 = (_Float16*)ws; ws += wbytes;
  _Float16* Qh   = (_Float16*)ws; ws += hbytes;
  _Float16* Kh   = (_Float16*)ws; ws += hbytes;
  _Float16* VhT  = (_Float16*)ws; ws += hbytes;
  _Float16* ctx  = (_Float16*)ws; ws += hbytes;      // total 72 MB

  cvt_f32_f16<<<512, 256, 0, stream>>>(wq, wq16, DM * DM);
  cvt_f32_f16<<<512, 256, 0, stream>>>(wk, wk16, DM * DM);
  cvt_f32_f16<<<512, 256, 0, stream>>>(wv, wv16, DM * DM);
  cvt_f32_f16<<<512, 256, 0, stream>>>(wo, wo16, DM * DM);

  const dim3 gproj(64, 16, 1);
  proj_gemm<0><<<gproj, 256, 0, stream>>>(q, nullptr, wq16, wqb, nullptr, Qh);
  proj_gemm<0><<<gproj, 256, 0, stream>>>(k, nullptr, wk16, wkb, nullptr, Kh);
  proj_gemm<1><<<gproj, 256, 0, stream>>>(v, nullptr, wv16, wvb, nullptr, VhT);

  attn_scores<<<dim3(Ss / 16, Hh, Bb), 256, 0, stream>>>(Qh, Kh, attn);
  attn_ctx<<<dim3(Ss / 16 / 8, Hh, Bb), 256, 0, stream>>>(attn, VhT, ctx);

  proj_gemm<2><<<gproj, 256, 0, stream>>>(nullptr, ctx, wo16, wob, out, nullptr);
}
